// STiT_10445360464485
// MI455X (gfx1250) — matrix operand order repaired and checked by reading, not compiled, not run on hardware
//
#include <hip/hip_runtime.h>

typedef _Float16 v16h __attribute__((ext_vector_type(16)));
typedef float    v8f  __attribute__((ext_vector_type(8)));

#define BATCH 64
#define DM    48
#define NP    3136
#define NC    3137
#define NH    4
#define IMGX  224
#define NEGV  (-987654321.0f)

// ---------------------------------------------------------------- init token
__global__ void k_init_token(const float* __restrict__ pt, float* __restrict__ tvec) {
  int i = blockIdx.x * blockDim.x + threadIdx.x;
  if (i < BATCH * DM) tvec[i] = pt[i % DM];
}

// ------------------------------------------------------- patchify x -> ctx
// ctx[b, ph*56+pw, c*16+py*4+px] = x[b,c,ph*4+py,pw*4+px]; vectorized float4.
__global__ void k_patchify(const float* __restrict__ x, float* __restrict__ ctx) {
  int i = blockIdx.x * blockDim.x + threadIdx.x;
  const int total = BATCH * NP * 12;
  if (i >= total) return;
  int q = i % 12;
  int p = (i / 12) % NP;
  int b = i / (12 * NP);
  int c = q >> 2, py = q & 3;
  int ph = p / 56, pw = p % 56;
  const float4 v = *(const float4*)(x + (((size_t)(b * 3 + c) * IMGX + ph * 4 + py) * IMGX + pw * 4));
  *(float4*)(ctx + ((size_t)b * NP + p) * DM + q * 4) = v;
}

// -------------------------------------------- per-layer token prep (per b)
// h = LN(t); q = h@Wq; wvec[h][c] = scale_h * sum_d Wkv_k[c, h*16+d] q[h*16+d]
__global__ void k_token_prep(const float* __restrict__ tvec,
                             const float* __restrict__ ln_g, const float* __restrict__ ln_b,
                             const float* __restrict__ Wq,   const float* __restrict__ Wkv,
                             const float* __restrict__ scale,
                             float* __restrict__ hbuf, float* __restrict__ wvec) {
  __shared__ float st[DM], sh[DM], sq[64], stats[2];
  int b = blockIdx.x, tid = threadIdx.x;
  if (tid < DM) st[tid] = tvec[b * DM + tid];
  __syncthreads();
  if (tid == 0) {
    float m = 0.f; for (int c = 0; c < DM; c++) m += st[c]; m /= DM;
    float v = 0.f; for (int c = 0; c < DM; c++) { float d = st[c] - m; v += d * d; } v /= DM;
    stats[0] = m; stats[1] = rsqrtf(v + 1e-5f);
  }
  __syncthreads();
  if (tid < DM) {
    float h = (st[tid] - stats[0]) * stats[1] * ln_g[tid] + ln_b[tid];
    sh[tid] = h; hbuf[b * DM + tid] = h;
  }
  __syncthreads();
  if (tid < 64) {
    float q = 0.f;
    for (int c = 0; c < DM; c++) q += sh[c] * Wq[c * 64 + tid];
    sq[tid] = q;
  }
  __syncthreads();
  if (tid < NH * DM) {
    int hh = tid / DM, c = tid % DM;
    float w = 0.f;
    for (int d = 0; d < 16; d++) w += Wkv[c * 128 + hh * 16 + d] * sq[hh * 16 + d];
    wvec[(b * NH + hh) * DM + c] = w * scale[hh];
  }
}

// ------------------------------------------- scores: s[b,h,j] = ctx_j . w_h
__global__ void k_scores(const float* __restrict__ ctx, const float* __restrict__ wvec,
                         float* __restrict__ sc) {
  __shared__ float tile[64 * DM];
  __shared__ float wsh[NH * DM];
  int b = blockIdx.x / 49, jt = blockIdx.x % 49, tid = threadIdx.x;
  int j0 = jt * 64;
  const float* cbase = ctx + ((size_t)b * NP + j0) * DM;
  for (int e = tid; e < 64 * DM; e += 256) tile[e] = cbase[e];
  if (tid < NH * DM) wsh[tid] = wvec[b * NH * DM + tid];
  __syncthreads();
  int jl = tid >> 2, hh = tid & 3;
  const float* tr = tile + jl * DM;
  const float* wr = wsh + hh * DM;
  float s = 0.f;
  #pragma unroll 8
  for (int c = 0; c < DM; c++) s += tr[c] * wr[c];
  sc[(size_t)(b * NH + hh) * NC + (j0 + jl + 1)] = s;
}

// -------------------------------- softmax with talking-heads pre/post mix
__device__ __forceinline__ void mix_dots(const float* __restrict__ sc, const float* pre,
                                         const float* colsum, int b, int j, float md[NH]) {
  if (j == 0) {
    #pragma unroll
    for (int g = 0; g < NH; g++) md[g] = NEGV * colsum[g];
  } else {
    float s[NH];
    #pragma unroll
    for (int h = 0; h < NH; h++) s[h] = sc[(size_t)(b * NH + h) * NC + j];
    #pragma unroll
    for (int g = 0; g < NH; g++)
      md[g] = s[0] * pre[g] + s[1] * pre[NH + g] + s[2] * pre[2 * NH + g] + s[3] * pre[3 * NH + g];
  }
}

__global__ void k_softmax(const float* __restrict__ sc, const float* __restrict__ mpre,
                          const float* __restrict__ mpost, float* __restrict__ at) {
  __shared__ float pre[16], post[16], colsum[NH], red[256], mg[NH], zg[NH];
  int b = blockIdx.x, tid = threadIdx.x;
  if (tid < 16) { pre[tid] = mpre[tid]; post[tid] = mpost[tid]; }
  __syncthreads();
  if (tid < NH) colsum[tid] = pre[tid] + pre[NH + tid] + pre[2 * NH + tid] + pre[3 * NH + tid];
  __syncthreads();
  float lm[NH] = {-1e30f, -1e30f, -1e30f, -1e30f};
  for (int j = tid; j < NC; j += 256) {
    float md[NH]; mix_dots(sc, pre, colsum, b, j, md);
    #pragma unroll
    for (int g = 0; g < NH; g++) lm[g] = fmaxf(lm[g], md[g]);
  }
  for (int g = 0; g < NH; g++) {
    red[tid] = lm[g]; __syncthreads();
    for (int s = 128; s > 0; s >>= 1) { if (tid < s) red[tid] = fmaxf(red[tid], red[tid + s]); __syncthreads(); }
    if (tid == 0) mg[g] = red[0];
    __syncthreads();
  }
  float lz[NH] = {0.f, 0.f, 0.f, 0.f};
  for (int j = tid; j < NC; j += 256) {
    float md[NH]; mix_dots(sc, pre, colsum, b, j, md);
    #pragma unroll
    for (int g = 0; g < NH; g++) lz[g] += expf(md[g] - mg[g]);
  }
  for (int g = 0; g < NH; g++) {
    red[tid] = lz[g]; __syncthreads();
    for (int s = 128; s > 0; s >>= 1) { if (tid < s) red[tid] += red[tid + s]; __syncthreads(); }
    if (tid == 0) zg[g] = red[0];
    __syncthreads();
  }
  for (int j = tid; j < NC; j += 256) {
    float md[NH]; mix_dots(sc, pre, colsum, b, j, md);
    float e[NH];
    #pragma unroll
    for (int g = 0; g < NH; g++) e[g] = expf(md[g] - mg[g]) / zg[g];
    #pragma unroll
    for (int g2 = 0; g2 < NH; g2++) {
      float a = e[0] * post[g2] + e[1] * post[NH + g2] + e[2] * post[2 * NH + g2] + e[3] * post[3 * NH + g2];
      at[(size_t)(b * NH + g2) * NC + j] = a;
    }
  }
}

// ------------------- weighted context sum, V-projection, Wo, residual add
__global__ void k_wsum(const float* __restrict__ ctx, const float* __restrict__ at,
                       const float* __restrict__ hbuf, const float* __restrict__ Wkv,
                       const float* __restrict__ Wo, const float* __restrict__ bo,
                       float* __restrict__ tvec) {
  __shared__ float tile[32 * DM];
  __shared__ float cv[NH * DM];
  __shared__ float o64[64];
  __shared__ float hsh[DM];
  int b = blockIdx.x, tid = threadIdx.x;
  if (tid < DM) hsh[tid] = hbuf[b * DM + tid];
  __syncthreads();
  int g = tid / DM, c = tid % DM;
  bool act = tid < NH * DM;
  const float* arow = at + (size_t)(b * NH + (act ? g : 0)) * NC;
  float acc = act ? arow[0] * hsh[c] : 0.f;     // j==0 term: attn(0) * h-token
  for (int t0 = 0; t0 < NP; t0 += 32) {
    const float* cb = ctx + ((size_t)b * NP + t0) * DM;
    for (int e = tid; e < 32 * DM; e += 256) tile[e] = cb[e];
    __builtin_prefetch(cb + 32 * DM, 0, 1);      // global_prefetch_b8, next tile
    __syncthreads();
    if (act) {
      #pragma unroll 4
      for (int jj = 0; jj < 32; jj++) acc += arow[t0 + jj + 1] * tile[jj * DM + c];
    }
    __syncthreads();
  }
  if (act) cv[tid] = acc;
  __syncthreads();
  if (tid < 64) {
    int gg = tid >> 4;
    float o = 0.f;
    for (int cc = 0; cc < DM; cc++) o += cv[gg * DM + cc] * Wkv[cc * 128 + 64 + tid];
    o64[tid] = o;
  }
  __syncthreads();
  if (tid < DM) {
    float v = bo[tid];
    for (int k = 0; k < 64; k++) v += o64[k] * Wo[k * DM + tid];
    tvec[b * DM + tid] += v;
  }
}

// ----------------------------------------------------------------- MLP (48)
__global__ void k_mlp(float* __restrict__ tvec, const float* __restrict__ g,
                      const float* __restrict__ bt, const float* __restrict__ W1,
                      const float* __restrict__ b1, const float* __restrict__ W2,
                      const float* __restrict__ b2) {
  __shared__ float st[DM], sh2[DM], su[DM], stats[2];
  int b = blockIdx.x, tid = threadIdx.x;
  if (tid < DM) st[tid] = tvec[b * DM + tid];
  __syncthreads();
  if (tid == 0) {
    float m = 0.f; for (int c = 0; c < DM; c++) m += st[c]; m /= DM;
    float v = 0.f; for (int c = 0; c < DM; c++) { float d = st[c] - m; v += d * d; } v /= DM;
    stats[0] = m; stats[1] = rsqrtf(v + 1e-5f);
  }
  __syncthreads();
  if (tid < DM) sh2[tid] = (st[tid] - stats[0]) * stats[1] * g[tid] + bt[tid];
  __syncthreads();
  if (tid < DM) {
    float u = b1[tid];
    for (int c = 0; c < DM; c++) u += sh2[c] * W1[c * DM + tid];
    su[tid] = 0.5f * u * (1.0f + erff(u * 0.70710678118654752f));   // exact GELU
  }
  __syncthreads();
  if (tid < DM) {
    float v = b2[tid];
    for (int c = 0; c < DM; c++) v += su[c] * W2[c * DM + tid];
    tvec[b * DM + tid] = st[tid] + v;
  }
}

// ---------------------------------------------------- head -> theta[b][24]
__global__ void k_head(const float* __restrict__ tvec, const float* __restrict__ g,
                       const float* __restrict__ bt, const float* __restrict__ hW,
                       const float* __restrict__ initp, float* __restrict__ theta) {
  __shared__ float st[DM], sh[DM], stats[2];
  int b = blockIdx.x, tid = threadIdx.x;
  if (tid < DM) st[tid] = tvec[b * DM + tid];
  __syncthreads();
  if (tid == 0) {
    float m = 0.f; for (int c = 0; c < DM; c++) m += st[c]; m /= DM;
    float v = 0.f; for (int c = 0; c < DM; c++) { float d = st[c] - m; v += d * d; } v /= DM;
    stats[0] = m; stats[1] = rsqrtf(v + 1e-5f);
  }
  __syncthreads();
  if (tid < DM) sh[tid] = (st[tid] - stats[0]) * stats[1] * g[tid] + bt[tid];
  __syncthreads();
  if (tid < 24) {
    float p = 0.f;
    for (int c = 0; c < DM; c++) p += sh[c] * hW[c * 24 + tid];
    theta[b * 24 + tid] = tanhf(p) + initp[tid];
  }
}

// ------- fused: affine warp sample + PatchMerging + LN + WMMA GEMM (60->96)
// 1 wave = 16 merged rows x all 96 outputs. 4 waves/block.
// B operand staged in LDS pre-packed in WMMA fragment order -> ds_load_b128.
// A operand read as contiguous float4 runs from in-place-normalized fl.
__global__ void __launch_bounds__(128)
k_final(const float* __restrict__ x, const float* __restrict__ theta,
        const float* __restrict__ png, const float* __restrict__ pnb,
        const float* __restrict__ pmW, float* __restrict__ out) {
  __shared__ float    fl[4][16 * 64];
  __shared__ _Float16 pmwf[12 * 32 * 16];       // [nt*2+s][lane][e] fragment-packed
  __shared__ float    rmean[4][16], rrstd[4][16];
  __shared__ float    thw[4][24];
  __shared__ float    lng[64], lnb[64];
  int tid = threadIdx.x;
  int wid = tid >> 5, lane = tid & 31;
  int gt = blockIdx.x * 4 + wid;          // global 16-row tile id
  int b = gt / 784, tl = gt % 784;
  // stage pm_W as f16 directly in WMMA B-fragment order (K padded 60->64)
  for (int e = tid; e < 12 * 32 * 16; e += 128) {
    int f  = e >> 9;            // 0..11 fragment (nt*2+s)
    int dl = (e >> 4) & 31;     // destination lane
    int el = e & 15;            // element within v16h
    int nt = f >> 1, s = f & 1;
    int k  = s * 32 + ((dl & 16) >> 1) + (el & 7) + ((el & 8) << 1);     // lanes16-31 -> K+8, el>=8 -> K+16
    int n  = nt * 16 + (dl & 15);
    pmwf[e] = (k < 60) ? (_Float16)pmW[k * 96 + n] : (_Float16)0.f;
  }
  if (tid < 60) { lng[tid] = png[tid]; lnb[tid] = pnb[tid]; }
  if (tid >= 60 && tid < 64) { lng[tid] = 0.f; lnb[tid] = 0.f; }
  if (lane < 24) thw[wid][lane] = theta[b * 24 + lane];
  __syncthreads();
  const float* xb = x + (size_t)b * 3 * 50176;
  // build 16 merged feature rows (60 feats each) in LDS
  for (int it = 0; it < 2; it++) {
    int task = it * 32 + lane;            // 64 (row, sub-pixel) tasks
    int r = task >> 2, pp = task & 3;
    int m  = tl * 16 + r;
    int mh = m / 112, mw = m % 112;
    int p1 = pp >> 1, p2 = pp & 1;
    int y = mh * 2 + p1, xc = mw * 2 + p2;
    int pix = y * IMGX + xc;
    float* fr = &fl[wid][r * 64 + pp * 15];
    fr[0] = xb[pix]; fr[1] = xb[50176 + pix]; fr[2] = xb[2 * 50176 + pix];
    float gx = (xc + 0.5f) * (2.0f / IMGX) - 1.0f;
    float gy = (y  + 0.5f) * (2.0f / IMGX) - 1.0f;
    #pragma unroll
    for (int i = 0; i < 4; i++) {
      const float* th = &thw[wid][i * 6];
      float tx = gx * th[0] + gy * th[1] + th[2];
      float ty = gx * th[3] + gy * th[4] + th[5];
      float ix = ((tx + 1.0f) * IMGX - 1.0f) * 0.5f;
      float iy = ((ty + 1.0f) * IMGX - 1.0f) * 0.5f;
      float x0f = floorf(ix), y0f = floorf(iy);
      float wx = ix - x0f, wy = iy - y0f;
      int x0 = (int)x0f, y0 = (int)y0f;
      bool vx0 = (x0 >= 0) & (x0 < IMGX),     vx1 = (x0 + 1 >= 0) & (x0 + 1 < IMGX);
      bool vy0 = (y0 >= 0) & (y0 < IMGX),     vy1 = (y0 + 1 >= 0) & (y0 + 1 < IMGX);
      int cx0 = min(max(x0, 0), IMGX - 1),    cx1 = min(max(x0 + 1, 0), IMGX - 1);
      int cy0 = min(max(y0, 0), IMGX - 1),    cy1 = min(max(y0 + 1, 0), IMGX - 1);
      int i00 = cy0 * IMGX + cx0, i01 = cy0 * IMGX + cx1;
      int i10 = cy1 * IMGX + cx0, i11 = cy1 * IMGX + cx1;
      float m00 = (vx0 && vy0) ? (1.f - wx) * (1.f - wy) : 0.f;
      float m01 = (vx1 && vy0) ? wx * (1.f - wy) : 0.f;
      float m10 = (vx0 && vy1) ? (1.f - wx) * wy : 0.f;
      float m11 = (vx1 && vy1) ? wx * wy : 0.f;
      #pragma unroll
      for (int ch = 0; ch < 3; ch++) {
        const float* xp = xb + ch * 50176;
        fr[3 + i * 3 + ch] = xp[i00] * m00 + xp[i01] * m01 + xp[i10] * m10 + xp[i11] * m11;
      }
    }
  }
  __syncthreads();
  // LayerNorm stats over the 60 features of each row
  if (lane < 16) {
    const float* row = &fl[wid][lane * 64];
    float mu = 0.f; for (int k = 0; k < 60; k++) mu += row[k]; mu /= 60.f;
    float va = 0.f; for (int k = 0; k < 60; k++) { float d = row[k] - mu; va += d * d; } va /= 60.f;
    rmean[wid][lane] = mu; rrstd[wid][lane] = rsqrtf(va + 1e-5f);
  }
  __syncthreads();
  // normalize fl IN PLACE (pad cols 60..63 with 0) so A-fragments are contiguous
  for (int e = lane; e < 16 * 64; e += 32) {
    int r = e >> 6, k = e & 63;
    fl[wid][e] = (k < 60) ? (fl[wid][e] - rmean[wid][r]) * rrstd[wid][r] * lng[k] + lnb[k] : 0.f;
  }
  __syncthreads();
  // pack A fragments (ISA 7.12.2 wave32 16-bit A 16x32 layout).
  // For lane (ml,hi) the fragment is two 8-float contiguous runs: base+{0..7,16..23}
  int ml = lane & 15, hi = lane >> 4;
  int base = ml * 64 + (hi ? 8 : 0);
  float4 p0 = *(const float4*)&fl[wid][base];
  float4 p1 = *(const float4*)&fl[wid][base + 4];
  float4 p2 = *(const float4*)&fl[wid][base + 16];
  float4 p3 = *(const float4*)&fl[wid][base + 20];
  float4 q0 = *(const float4*)&fl[wid][base + 32];
  float4 q1 = *(const float4*)&fl[wid][base + 36];
  float4 q2 = *(const float4*)&fl[wid][base + 48];
  float4 q3 = *(const float4*)&fl[wid][base + 52];
  float af0[16] = {p0.x,p0.y,p0.z,p0.w, p1.x,p1.y,p1.z,p1.w,
                   p2.x,p2.y,p2.z,p2.w, p3.x,p3.y,p3.z,p3.w};
  float af1[16] = {q0.x,q0.y,q0.z,q0.w, q1.x,q1.y,q1.z,q1.w,
                   q2.x,q2.y,q2.z,q2.w, q3.x,q3.y,q3.z,q3.w};
  v16h a0, a1;
  #pragma unroll
  for (int e = 0; e < 16; e++) { a0[e] = (_Float16)af0[e]; a1[e] = (_Float16)af1[e]; }
  size_t orow = (size_t)b * 12544 + (size_t)tl * 16;
  #pragma unroll
  for (int nt = 0; nt < 6; nt++) {
    v16h b0 = *(const v16h*)&pmwf[((nt * 2 + 0) * 32 + lane) * 16];   // 32B LDS vector loads
    v16h b1 = *(const v16h*)&pmwf[((nt * 2 + 1) * 32 + lane) * 16];
    v8f acc = {0.f, 0.f, 0.f, 0.f, 0.f, 0.f, 0.f, 0.f};
    acc = __builtin_amdgcn_wmma_f32_16x16x32_f16(false, a0, false, b0, (short)0, acc, false, false);
    acc = __builtin_amdgcn_wmma_f32_16x16x32_f16(false, a1, false, b1, (short)0, acc, false, false);
    #pragma unroll
    for (int r8 = 0; r8 < 8; r8++) {       // D: VGPR r -> (M=r / r+8, N=lane)
      int M = r8 + (hi ? 8 : 0);
      out[(orow + M) * 96 + nt * 16 + ml] = acc[r8];
    }
  }
}

// ----------------------------------------------------------------- launch
extern "C" void kernel_launch(void* const* d_in, const int* in_sizes, int n_in,
                              void* d_out, int out_size, void* d_ws, size_t ws_size,
                              hipStream_t stream) {
  const float* x      = (const float*)d_in[0];
  const float* ptok   = (const float*)d_in[1];
  const float* initp  = (const float*)d_in[2];
  const float* ln1_g  = (const float*)d_in[3];
  const float* ln1_b  = (const float*)d_in[4];
  const float* Wq     = (const float*)d_in[5];
  const float* Wkv    = (const float*)d_in[6];
  const float* ascale = (const float*)d_in[7];
  const float* mpre   = (const float*)d_in[8];
  const float* mpost  = (const float*)d_in[9];
  const float* Wo     = (const float*)d_in[10];
  const float* bo     = (const float*)d_in[11];
  const float* ln2_g  = (const float*)d_in[12];
  const float* ln2_b  = (const float*)d_in[13];
  const float* W1     = (const float*)d_in[14];
  const float* b1     = (const float*)d_in[15];
  const float* W2     = (const float*)d_in[16];
  const float* b2     = (const float*)d_in[17];
  const float* hlg    = (const float*)d_in[18];
  const float* hlb    = (const float*)d_in[19];
  const float* headW  = (const float*)d_in[20];
  const float* png    = (const float*)d_in[21];
  const float* pnb    = (const float*)d_in[22];
  const float* pmW    = (const float*)d_in[23];
  float* out = (float*)d_out;

  // workspace layout (floats), ~45 MB total
  float* ws   = (float*)d_ws;
  float* ctx  = ws;                                  // 64*3136*48
  float* tvec = ctx  + (size_t)BATCH * NP * DM;      // 64*48
  float* hbuf = tvec + BATCH * DM;                   // 64*48
  float* wvec = hbuf + BATCH * DM;                   // 64*4*48
  float* sc   = wvec + BATCH * NH * DM;              // 64*4*3137
  float* at   = sc   + (size_t)BATCH * NH * NC;      // 64*4*3137
  float* th   = at   + (size_t)BATCH * NH * NC;      // 64*24

  k_init_token<<<(BATCH * DM + 255) / 256, 256, 0, stream>>>(ptok, tvec);
  k_patchify<<<(BATCH * NP * 12 + 255) / 256, 256, 0, stream>>>(x, ctx);

  for (int l = 0; l < 2; l++) {
    k_token_prep<<<BATCH, 192, 0, stream>>>(tvec, ln1_g + l * DM, ln1_b + l * DM,
                                            Wq + l * DM * 64, Wkv + l * DM * 128,
                                            ascale + l * NH, hbuf, wvec);
    k_scores<<<BATCH * 49, 256, 0, stream>>>(ctx, wvec, sc);
    k_softmax<<<BATCH, 256, 0, stream>>>(sc, mpre + l * 16, mpost + l * 16, at);
    k_wsum<<<BATCH, 256, 0, stream>>>(ctx, at, hbuf, Wkv + l * DM * 128,
                                      Wo + l * 64 * DM, bo + l * DM, tvec);
    k_mlp<<<BATCH, 64, 0, stream>>>(tvec, ln2_g + l * DM, ln2_b + l * DM,
                                    W1 + l * DM * DM, b1 + l * DM,
                                    W2 + l * DM * DM, b2 + l * DM);
  }
  k_head<<<BATCH, 64, 0, stream>>>(tvec, hlg, hlb, headW, initp, th);
  k_final<<<(BATCH * 784) / 4, 128, 0, stream>>>(x, th, png, pnb, pmW, out);
}
